// AttGNN_9036611191117
// MI455X (gfx1250) — matrix operand order repaired and checked by reading, not compiled, not run on hardware
//
#include <hip/hip_runtime.h>
#include <hip/hip_bf16.h>

typedef __attribute__((ext_vector_type(16))) _Float16 v16h;
typedef __attribute__((ext_vector_type(8)))  float    v8f;

#define DH 128
#define TPB 256

__device__ __forceinline__ float lrelu01(float x) { return x > 0.f ? x : 0.01f * x; }

union FragU { v16h v; _Float16 h[16]; unsigned u[8]; };

// ---------------- utility fills ----------------
__global__ void fill_u32_kernel(unsigned* __restrict__ p, unsigned val, int n) {
  int i = blockIdx.x * blockDim.x + threadIdx.x;
  if (i < n) p[i] = val;
}

__global__ void init_bias_kernel(float* __restrict__ out, const float* __restrict__ bias, int total) {
  int i = blockIdx.x * blockDim.x + threadIdx.x;
  if (i < total) out[i] = bias[i & (DH - 1)];
}

// ---------------- f32 -> f16 conversion (vectorized, n % 4 == 0) ----------------
__global__ void cvt_f16_kernel(const float* __restrict__ in, _Float16* __restrict__ out, long n) {
  long i = ((long)blockIdx.x * blockDim.x + threadIdx.x) * 4;
  if (i >= n) return;
  float4 v = *(const float4*)&in[i];
  union { _Float16 h[4]; uint2 u; } r;
  r.h[0] = (_Float16)v.x; r.h[1] = (_Float16)v.y;
  r.h[2] = (_Float16)v.z; r.h[3] = (_Float16)v.w;
  *(uint2*)&out[i] = r.u;
}

// B [K, Ncols] f32 row-major  ->  Bt [Ncols, K] f16 (column-major of B)
__global__ void cvtT_f16_kernel(const float* __restrict__ B, _Float16* __restrict__ Bt,
                                int K, int Ncols) {
  int i = blockIdx.x * blockDim.x + threadIdx.x;
  if (i >= K * Ncols) return;
  int k = i / Ncols, n = i - k * Ncols;
  Bt[(long)n * K + k] = (_Float16)B[i];
}

// ---------------- WMMA GEMM: C[M,Ncols] = act(A16[M,K] @ Bt16[Ncols,K]^T + bias) ----------------
// block = 256 threads = 8 waves; wave w computes the 16x16 tile at cols [colblk+16w, +16).
// Fragments loaded straight from global memory in contiguous k order (lane half h holds k 16h..16h+15): 2x b128 per operand per K-step.
// gridDim.x = ceil(M/16), gridDim.y = ceil(Ncols/128). K must be a multiple of 32.
__global__ void gemm_wmma_kernel(const _Float16* __restrict__ A16,
                                 const _Float16* __restrict__ Bt16,
                                 float* __restrict__ C, int M, int K, int Ncols,
                                 int ldc, int col_off, const float* __restrict__ bias, int act) {
  const int tid  = threadIdx.x;
  const int w    = tid >> 5;
  const int lane = tid & 31;
  const int row0   = blockIdx.x * 16;
  const int colblk = blockIdx.y * 128;

  const int mA  = lane & 15;
  const int kbA = (lane >> 4) << 4;        // 0 or 16 (halves)
  const int nB  = colblk + (w << 4) + (lane & 15);
  const int kbB = (lane >> 4) << 4;        // 0 or 16 (halves)

  // clamp out-of-range lanes to a valid address; their results stay confined to
  // their own output row/col (WMMA lane ownership) and are masked at the store.
  int rowA = row0 + mA;   if (rowA >= M)     rowA = M - 1;
  int colB = nB;          if (colB >= Ncols) colB = Ncols - 1;

  const uint4* pa = (const uint4*)(A16  + (long)rowA * K + kbA);
  const uint4* pb = (const uint4*)(Bt16 + (long)colB * K + kbB);

  v8f acc = {};
#pragma unroll 2
  for (int k0 = 0; k0 < K; k0 += 32) {
    __builtin_prefetch(pa + 16, 0, 3);     // 256B ahead in this lane's A row
    __builtin_prefetch(pb + 16, 0, 3);
    uint4 a0 = pa[0];                      // halves kbA + 0..7
    uint4 a1 = pa[1];                      // halves kbA + 8..15
    uint4 b0 = pb[0];                      // halves kbB + 0..7
    uint4 b1 = pb[1];                      // halves kbB + 8..15
    FragU af, bf;
    af.u[0] = a0.x; af.u[1] = a0.y; af.u[2] = a0.z; af.u[3] = a0.w;
    af.u[4] = a1.x; af.u[5] = a1.y; af.u[6] = a1.z; af.u[7] = a1.w;
    bf.u[0] = b0.x; bf.u[1] = b0.y; bf.u[2] = b0.z; bf.u[3] = b0.w;
    bf.u[4] = b1.x; bf.u[5] = b1.y; bf.u[6] = b1.z; bf.u[7] = b1.w;
    acc = __builtin_amdgcn_wmma_f32_16x16x32_f16(false, af.v, false, bf.v,
                                                 (short)0, acc, false, false);
    pa += 4;                               // +32 halves
    pb += 4;
  }

#pragma unroll
  for (int r = 0; r < 8; ++r) {
    int row = row0 + r + ((lane >> 4) << 3);
    int col = colblk + (w << 4) + (lane & 15);
    if (row < M && col < Ncols) {
      float v = acc[r];
      if (bias) v += bias[col];
      if (act == 1) v = lrelu01(v);
      C[(long)row * ldc + col_off + col] = v;
    }
  }
}

// ---------------- per-node attention scalars: a_s = h.att_src, a_d = h.att_dst ----------------
__global__ void att_kernel(const float* __restrict__ h, const float* __restrict__ att_s,
                           const float* __restrict__ att_d, float* __restrict__ a_s,
                           float* __restrict__ a_d, int N) {
  int gw = (blockIdx.x * blockDim.x + threadIdx.x) >> 5;
  int lane = threadIdx.x & 31;
  if (gw >= N) return;
  float4 hv  = *(const float4*)&h[(long)gw * DH + lane * 4];
  float4 asv = *(const float4*)&att_s[lane * 4];
  float4 adv = *(const float4*)&att_d[lane * 4];
  float ps = hv.x * asv.x + hv.y * asv.y + hv.z * asv.z + hv.w * asv.w;
  float pd = hv.x * adv.x + hv.y * adv.y + hv.z * adv.z + hv.w * adv.w;
  for (int m = 16; m >= 1; m >>= 1) {
    ps += __shfl_xor(ps, m, 32);
    pd += __shfl_xor(pd, m, 32);
  }
  if (lane == 0) { a_s[gw] = ps; a_d[gw] = pd; }
}

// ordered-uint encoding for float atomic max
__device__ __forceinline__ unsigned fenc(float f) {
  unsigned u = __float_as_uint(f);
  return (u & 0x80000000u) ? ~u : (u | 0x80000000u);
}
__device__ __forceinline__ float fdec(unsigned u) {
  return __uint_as_float((u & 0x80000000u) ? (u ^ 0x80000000u) : ~u);
}

__global__ void edge_max_kernel(const int* __restrict__ src, const int* __restrict__ dst,
                                const float* __restrict__ a_s, const float* __restrict__ a_d,
                                unsigned* __restrict__ mbits, int E, int N) {
  int i = blockIdx.x * blockDim.x + threadIdx.x;
  if (i >= E + N) return;
  int s = (i < E) ? src[i] : (i - E);
  int d = (i < E) ? dst[i] : (i - E);
  float e = a_s[s] + a_d[d];
  e = e > 0.f ? e : 0.2f * e;   // leaky_relu(0.2)
  atomicMax(&mbits[d], fenc(e));
}

__global__ void decode_m_kernel(unsigned* __restrict__ mb, int N) {
  int i = blockIdx.x * blockDim.x + threadIdx.x;
  if (i >= N) return;
  float v = fdec(mb[i]);
  if ((__float_as_uint(v) & 0x7F800000u) == 0x7F800000u) v = 0.f;  // isfinite fixup
  ((float*)mb)[i] = v;
}

__global__ void edge_exp_kernel(const int* __restrict__ src, const int* __restrict__ dst,
                                const float* __restrict__ a_s, const float* __restrict__ a_d,
                                const float* __restrict__ m, float* __restrict__ p,
                                float* __restrict__ ssum, int E, int N) {
  int i = blockIdx.x * blockDim.x + threadIdx.x;
  if (i >= E + N) return;
  int s = (i < E) ? src[i] : (i - E);
  int d = (i < E) ? dst[i] : (i - E);
  float e = a_s[s] + a_d[d];
  e = e > 0.f ? e : 0.2f * e;
  float pe = __expf(e - m[d]);
  p[i] = pe;
  atomicAdd(&ssum[d], pe);
}

// wave per edge: out[dst] += alpha * h[src]; lane covers 4 consecutive features
__global__ void edge_scatter_kernel(const int* __restrict__ src, const int* __restrict__ dst,
                                    const float* __restrict__ p, const float* __restrict__ ssum,
                                    const float* __restrict__ h, float* __restrict__ out,
                                    int E, int N) {
  long gw = (long)blockIdx.x * (blockDim.x >> 5) + (threadIdx.x >> 5);
  int lane = threadIdx.x & 31;
  if (gw >= (long)E + N) return;
  int s = (gw < E) ? src[gw] : (int)(gw - E);
  int d = (gw < E) ? dst[gw] : (int)(gw - E);
  float alpha = p[gw] / ssum[d];
  float4 hv = *(const float4*)&h[(long)s * DH + lane * 4];
  float* o = &out[(long)d * DH + lane * 4];
  atomicAdd(o + 0, alpha * hv.x);
  atomicAdd(o + 1, alpha * hv.y);
  atomicAdd(o + 2, alpha * hv.z);
  atomicAdd(o + 3, alpha * hv.w);
}

// wave per node: pool[batch[n]] += f(in[n]); cnt[batch[n]] += 1
__global__ void pool_acc_kernel(const float* __restrict__ in, const int* __restrict__ batch,
                                float* __restrict__ pool, float* __restrict__ cnt, int N, int mode) {
  int gw = (blockIdx.x * blockDim.x + threadIdx.x) >> 5;
  int lane = threadIdx.x & 31;
  if (gw >= N) return;
  int g = batch[gw];
  float4 v = *(const float4*)&in[(long)gw * DH + lane * 4];
  if (mode == 1) { v.x = lrelu01(v.x); v.y = lrelu01(v.y); v.z = lrelu01(v.z); v.w = lrelu01(v.w); }
  float* o = &pool[(long)g * DH + lane * 4];
  atomicAdd(o + 0, v.x);
  atomicAdd(o + 1, v.y);
  atomicAdd(o + 2, v.z);
  atomicAdd(o + 3, v.w);
  if (lane == 0) atomicAdd(&cnt[g], 1.0f);
}

__global__ void pool_fin_kernel(const float* __restrict__ pool, const float* __restrict__ cnt,
                                float* __restrict__ out, int ldc, int col_off, int act, int G) {
  int i = blockIdx.x * blockDim.x + threadIdx.x;
  if (i >= G * DH) return;
  int g = i >> 7, f = i & 127;
  float c = cnt[g]; if (c < 1.f) c = 1.f;
  float v = pool[i] / c;
  if (act) v = lrelu01(v);
  out[(long)g * ldc + col_off + f] = v;
}

__global__ void final_out_kernel(const float* __restrict__ y, const float* __restrict__ w,
                                 const float* __restrict__ b, float* __restrict__ out, int G) {
  int g = threadIdx.x;
  if (g >= G) return;
  float acc = b[0];
  for (int j = 0; j < 64; ++j) acc += y[g * 64 + j] * w[j];
  out[g] = 1.f / (1.f + __expf(-acc));
}

// ---------------- host-side orchestration ----------------
extern "C" void kernel_launch(void* const* d_in, const int* in_sizes, int n_in,
                              void* d_out, int out_size, void* d_ws, size_t ws_size,
                              hipStream_t stream) {
  const float* x1  = (const float*)d_in[0];
  const int*   ei1 = (const int*)  d_in[1];
  const int*   bt1 = (const int*)  d_in[2];
  const float* x2  = (const float*)d_in[3];
  const int*   ei2 = (const int*)  d_in[4];
  const int*   bt2 = (const int*)  d_in[5];
  const float* W1  = (const float*)d_in[6];
  const float* atS1= (const float*)d_in[7];
  const float* atD1= (const float*)d_in[8];
  const float* bg1 = (const float*)d_in[9];
  const float* W2  = (const float*)d_in[10];
  const float* atS2= (const float*)d_in[11];
  const float* atD2= (const float*)d_in[12];
  const float* bg2 = (const float*)d_in[13];
  const float* fw1 = (const float*)d_in[14];
  const float* fb1 = (const float*)d_in[15];
  const float* fw2 = (const float*)d_in[16];
  const float* fb2 = (const float*)d_in[17];
  const float* fc1w= (const float*)d_in[18];
  const float* fc1b= (const float*)d_in[19];
  const float* fc2w= (const float*)d_in[20];
  const float* fc2b= (const float*)d_in[21];
  const float* outw= (const float*)d_in[22];
  const float* outb= (const float*)d_in[23];

  const int DIN = 1024;
  const int N = in_sizes[0] / DIN;
  const int E = in_sizes[1] / 2;
  const int G = in_sizes[21];   // fc2b size == 64
  const int EN = E + N;

  const int* src1 = ei1;       const int* dst1 = ei1 + E;
  const int* src2 = ei2;       const int* dst2 = ei2 + E;

  // workspace carve-out (256B aligned)
  char* wp = (char*)d_ws;
  auto alloc = [&](size_t bytes) -> void* {
    void* r = (void*)wp;
    wp += (bytes + 255) & ~(size_t)255;
    return r;
  };
  float* h1   = (float*)alloc((size_t)N * DH * 4);
  float* h2   = (float*)alloc((size_t)N * DH * 4);
  float* agg1 = (float*)alloc((size_t)N * DH * 4);
  float* agg2 = (float*)alloc((size_t)N * DH * 4);
  float* z2   = (float*)alloc((size_t)N * DH * 4);
  float* aS1  = (float*)alloc((size_t)N * 4);
  float* aD1  = (float*)alloc((size_t)N * 4);
  float* aS2  = (float*)alloc((size_t)N * 4);
  float* aD2  = (float*)alloc((size_t)N * 4);
  float* m1   = (float*)alloc((size_t)N * 4);
  float* m2   = (float*)alloc((size_t)N * 4);
  float* s1   = (float*)alloc((size_t)N * 4);
  float* s2   = (float*)alloc((size_t)N * 4);
  float* p1   = (float*)alloc((size_t)EN * 4);
  float* p2   = (float*)alloc((size_t)EN * 4);
  float* pool1   = (float*)alloc((size_t)G * DH * 4);
  float* pool2   = (float*)alloc((size_t)G * DH * 4);
  float* pooled1 = (float*)alloc((size_t)G * DH * 4);
  float* cnt1 = (float*)alloc((size_t)G * 4);
  float* cnt2 = (float*)alloc((size_t)G * 4);
  float* xc = (float*)alloc((size_t)G * 256 * 4);
  float* y1 = (float*)alloc((size_t)G * 256 * 4);
  float* y2 = (float*)alloc((size_t)G * 64 * 4);
  // f16 staging buffers (reused sequentially; stream-ordered so reuse is safe)
  _Float16* xa16 = (_Float16*)alloc((size_t)N * DIN * 2);  // activations, K=1024
  _Float16* as16 = (_Float16*)alloc((size_t)N * DH  * 2);  // activations, K<=256
  _Float16* bt16 = (_Float16*)alloc((size_t)DIN * DH * 2); // transposed weights (max 256KB)

  const unsigned ENC_NEG_INF = 0x007FFFFFu;  // fenc(-inf)
  dim3 blk(TPB);
  int nbN   = (N + TPB - 1) / TPB;
  int nbE   = (EN + TPB - 1) / TPB;
  int nbNW  = (N + 7) / 8;       // wave-per-node kernels
  int nbEW  = (EN + 7) / 8;      // wave-per-edge kernels
  int nbND  = (N * DH + TPB - 1) / TPB;
  int nbGD  = (G * DH + TPB - 1) / TPB;

  long nX = (long)N * DIN;
  int nbX4  = (int)((nX / 4 + TPB - 1) / TPB);
  int nbH4  = (int)(((long)N * DH / 4 + TPB - 1) / TPB);

  // ===== branch 1: GATConv =====
  cvt_f16_kernel<<<nbX4, blk, 0, stream>>>(x1, xa16, nX);
  cvtT_f16_kernel<<<(DIN * DH + TPB - 1) / TPB, blk, 0, stream>>>(W1, bt16, DIN, DH);
  gemm_wmma_kernel<<<dim3((N + 15) / 16, 1), blk, 0, stream>>>(xa16, bt16, h1, N, DIN, DH, DH, 0, nullptr, 0);
  att_kernel<<<nbNW, blk, 0, stream>>>(h1, atS1, atD1, aS1, aD1, N);
  fill_u32_kernel<<<nbN, blk, 0, stream>>>((unsigned*)m1, ENC_NEG_INF, N);
  fill_u32_kernel<<<nbN, blk, 0, stream>>>((unsigned*)s1, 0u, N);
  edge_max_kernel<<<nbE, blk, 0, stream>>>(src1, dst1, aS1, aD1, (unsigned*)m1, E, N);
  decode_m_kernel<<<nbN, blk, 0, stream>>>((unsigned*)m1, N);
  edge_exp_kernel<<<nbE, blk, 0, stream>>>(src1, dst1, aS1, aD1, m1, p1, s1, E, N);
  init_bias_kernel<<<nbND, blk, 0, stream>>>(agg1, bg1, N * DH);
  edge_scatter_kernel<<<nbEW, blk, 0, stream>>>(src1, dst1, p1, s1, h1, agg1, E, N);
  // pool(mean of lrelu) then fc+lrelu into xc[:, 0:128]
  fill_u32_kernel<<<nbGD, blk, 0, stream>>>((unsigned*)pool1, 0u, G * DH);
  fill_u32_kernel<<<1, blk, 0, stream>>>((unsigned*)cnt1, 0u, G);
  pool_acc_kernel<<<nbNW, blk, 0, stream>>>(agg1, bt1, pool1, cnt1, N, 1);
  pool_fin_kernel<<<nbGD, blk, 0, stream>>>(pool1, cnt1, pooled1, DH, 0, 0, G);
  cvt_f16_kernel<<<(G * DH / 4 + TPB - 1) / TPB, blk, 0, stream>>>(pooled1, as16, (long)G * DH);
  cvtT_f16_kernel<<<(DH * DH + TPB - 1) / TPB, blk, 0, stream>>>(fw1, bt16, DH, DH);
  gemm_wmma_kernel<<<dim3((G + 15) / 16, 1), blk, 0, stream>>>(as16, bt16, xc, G, DH, DH, 256, 0, fb1, 1);

  // ===== branch 2: GATConv =====
  cvt_f16_kernel<<<nbX4, blk, 0, stream>>>(x2, xa16, nX);
  cvtT_f16_kernel<<<(DIN * DH + TPB - 1) / TPB, blk, 0, stream>>>(W2, bt16, DIN, DH);
  gemm_wmma_kernel<<<dim3((N + 15) / 16, 1), blk, 0, stream>>>(xa16, bt16, h2, N, DIN, DH, DH, 0, nullptr, 0);
  att_kernel<<<nbNW, blk, 0, stream>>>(h2, atS2, atD2, aS2, aD2, N);
  fill_u32_kernel<<<nbN, blk, 0, stream>>>((unsigned*)m2, ENC_NEG_INF, N);
  fill_u32_kernel<<<nbN, blk, 0, stream>>>((unsigned*)s2, 0u, N);
  edge_max_kernel<<<nbE, blk, 0, stream>>>(src2, dst2, aS2, aD2, (unsigned*)m2, E, N);
  decode_m_kernel<<<nbN, blk, 0, stream>>>((unsigned*)m2, N);
  edge_exp_kernel<<<nbE, blk, 0, stream>>>(src2, dst2, aS2, aD2, m2, p2, s2, E, N);
  init_bias_kernel<<<nbND, blk, 0, stream>>>(agg2, bg2, N * DH);
  edge_scatter_kernel<<<nbEW, blk, 0, stream>>>(src2, dst2, p2, s2, h2, agg2, E, N);
  // per-node fc+lrelu, then mean-pool, then lrelu into xc[:, 128:256]
  cvt_f16_kernel<<<nbH4, blk, 0, stream>>>(agg2, as16, (long)N * DH);
  cvtT_f16_kernel<<<(DH * DH + TPB - 1) / TPB, blk, 0, stream>>>(fw2, bt16, DH, DH);
  gemm_wmma_kernel<<<dim3((N + 15) / 16, 1), blk, 0, stream>>>(as16, bt16, z2, N, DH, DH, DH, 0, fb2, 1);
  fill_u32_kernel<<<nbGD, blk, 0, stream>>>((unsigned*)pool2, 0u, G * DH);
  fill_u32_kernel<<<1, blk, 0, stream>>>((unsigned*)cnt2, 0u, G);
  pool_acc_kernel<<<nbNW, blk, 0, stream>>>(z2, bt2, pool2, cnt2, N, 0);
  pool_fin_kernel<<<nbGD, blk, 0, stream>>>(pool2, cnt2, xc, 256, 128, 1, G);

  // ===== head =====
  cvt_f16_kernel<<<(G * 256 / 4 + TPB - 1) / TPB, blk, 0, stream>>>(xc, as16, (long)G * 256);
  cvtT_f16_kernel<<<(256 * 256 + TPB - 1) / TPB, blk, 0, stream>>>(fc1w, bt16, 256, 256);
  gemm_wmma_kernel<<<dim3((G + 15) / 16, 2), blk, 0, stream>>>(as16, bt16, y1, G, 256, 256, 256, 0, fc1b, 1);
  cvt_f16_kernel<<<(G * 256 / 4 + TPB - 1) / TPB, blk, 0, stream>>>(y1, as16, (long)G * 256);
  cvtT_f16_kernel<<<(256 * 64 + TPB - 1) / TPB, blk, 0, stream>>>(fc2w, bt16, 256, 64);
  gemm_wmma_kernel<<<dim3((G + 15) / 16, 1), blk, 0, stream>>>(as16, bt16, y2, G, 256, 64, 64, 0, fc2b, 1);
  final_out_kernel<<<1, 64, 0, stream>>>(y2, outw, outb, (float*)d_out, G);
}
